// Model_46239617908980
// MI455X (gfx1250) — matrix operand order repaired and checked by reading, not compiled, not run on hardware
//
#include <hip/hip_runtime.h>
#include <math.h>

// ---------------------------------------------------------------------------
// Model dims (fixed by the reference)
// ---------------------------------------------------------------------------
#define Bc   4
#define Sc   1024
#define Dc   512
#define Hc   8
#define Lc   6
#define Wc   64
#define Oc   300
#define DHc  64          // D / H
#define Rc   131         // 2W + 3
#define EPSc 1e-6f

// ---------------------------------------------------------------------------
// CDNA5 WMMA types (wave32, 16x16x32 bf16 -> f32)
// ---------------------------------------------------------------------------
typedef __bf16 bf16;
typedef __attribute__((ext_vector_type(2)))  bf16  v2bf;
typedef __attribute__((ext_vector_type(8)))  bf16  v8bf;
typedef __attribute__((ext_vector_type(16))) bf16  v16bf;
typedef __attribute__((ext_vector_type(4)))  float v4f;
typedef __attribute__((ext_vector_type(8)))  float v8f;

__device__ __forceinline__ v16bf concat8(v8bf lo, v8bf hi) {
  return __builtin_shufflevector(lo, hi, 0,1,2,3,4,5,6,7,8,9,10,11,12,13,14,15);
}
__device__ __forceinline__ v4f load4(const float* p) {
  return *reinterpret_cast<const v4f*>(p);
}
__device__ __forceinline__ v8bf pack8(v4f a, v4f b) {
  v8bf p;
  #pragma unroll
  for (int i = 0; i < 4; ++i) { p[i] = (bf16)a[i]; p[i + 4] = (bf16)b[i]; }
  return p;
}

// ---------------------------------------------------------------------------
// Generic batched WMMA GEMM:
//   C[z] = act( alpha * A[z] @ (BT ? B[z]^T : B[z]) + bias ) + res[z]
// A: [M,K] row-major (ld=K). B: BT==0 -> [K,N] (ld=N); BT==1 -> [N,K] (ld=K).
// C/res: [M,N] (ld=N). Tile 128x64, K-step 32, 256 threads = 8 wave32 waves.
// Wave w owns m-band [w*16, w*16+16); one A fragment feeds 4 v_wmma / K-step.
// OOB rows/cols are address-CLAMPED on load (duplicate data, no exec masking);
// epilogue store guards discard those outputs. Requires K % 32 == 0 (all call
// sites satisfy this). Global loads for step k+1 are issued while step k's
// WMMAs run (software pipelining through VGPR staging).
// ---------------------------------------------------------------------------
#define TILE_M 128
#define TILE_N 64
#define TILE_K 32
#define LDS_LD (TILE_K + 8)   // 40 bf16 = 80B row stride; keeps 16B alignment

template <int BT>
__global__ __launch_bounds__(256)
void wmma_gemm_kernel(const float* __restrict__ A, const float* __restrict__ B,
                      const float* __restrict__ bias, const float* __restrict__ res,
                      float* __restrict__ C,
                      int M, int N, int K,
                      long long strideA, long long strideB, long long strideC,
                      float alpha, int act)
{
  __shared__ bf16 As[TILE_M][LDS_LD];
  __shared__ bf16 Bs[TILE_N][LDS_LD];   // [n][k]: fragment reads contiguous in k

  const int tid  = threadIdx.x;
  const int wave = tid >> 5;
  const int lane = tid & 31;
  const int half = lane >> 4;   // K-group selector within a fragment
  const int l16  = lane & 15;

  const int m0 = blockIdx.y * TILE_M;
  const int n0 = blockIdx.x * TILE_N;
  const long long z = blockIdx.z;

  A += z * strideA;
  B += z * strideB;
  C += z * strideC;
  const float* resz = res ? (res + z * strideC) : nullptr;

  const int mrow = wave * 16 + l16;   // A-fragment row (lanes 0-15 == 16-31)

  v8f acc[4];
  #pragma unroll
  for (int t = 0; t < 4; ++t) acc[t] = v8f{0.f,0.f,0.f,0.f,0.f,0.f,0.f,0.f};

  // ---- cooperative-load index precompute (clamped addresses, no guards) ----
  const int ar  = tid >> 1;                 // A row 0..127
  const int ac  = (tid & 1) * 16;           // A col 0 or 16
  const int car = (m0 + ar < M) ? (m0 + ar) : (M - 1);
  const float* aRow = A + (long long)car * K + ac;

  // BT path: n = tid>>2 (0..63), k col (tid&3)*8
  const int tr  = tid >> 2;
  const int tc  = (tid & 3) * 8;
  const int ctr = (n0 + tr < N) ? (n0 + tr) : (N - 1);
  const float* bRowT = B + (long long)ctr * K + tc;

  // B[K,N] path: even k pair (tid&15)*2, n group (tid>>4)*4
  const int bkk = (tid & 15) * 2;
  const int bnb = (tid >> 4) * 4;
  const bool bFast = (n0 + bnb + 3 < N) && ((N & 3) == 0);

  // ---- VGPR staging for software pipeline ----
  v4f a0, a1, a2, a3;       // A: 16 floats
  v4f b0v, b1v;             // B: 8 floats (BT) or two 4-float k-rows (B[K,N])

  auto fetch = [&](int k0) {
    a0 = load4(aRow + k0);  a1 = load4(aRow + k0 + 4);
    a2 = load4(aRow + k0 + 8); a3 = load4(aRow + k0 + 12);
    if (BT) {
      b0v = load4(bRowT + k0); b1v = load4(bRowT + k0 + 4);
    } else {
      const float* bp0 = B + (long long)(k0 + bkk) * N + n0 + bnb;
      const float* bp1 = bp0 + N;
      if (bFast) {
        b0v = load4(bp0); b1v = load4(bp1);
      } else {   // ragged/odd N: clamped scalar loads (no exec masking)
        #pragma unroll
        for (int i = 0; i < 4; ++i) {
          const int gn = n0 + bnb + i;
          const int adj = (gn < N) ? 0 : (N - 1 - gn);   // clamp to last col
          b0v[i] = bp0[i + adj];
          b1v[i] = bp1[i + adj];
        }
      }
    }
  };

  auto stage = [&]() {
    *reinterpret_cast<v8bf*>(&As[ar][ac])     = pack8(a0, a1);
    *reinterpret_cast<v8bf*>(&As[ar][ac + 8]) = pack8(a2, a3);
    if (BT) {
      *reinterpret_cast<v8bf*>(&Bs[tr][tc]) = pack8(b0v, b1v);
    } else {
      #pragma unroll
      for (int i = 0; i < 4; ++i) {
        v2bf p; p[0] = (bf16)b0v[i]; p[1] = (bf16)b1v[i];
        *reinterpret_cast<v2bf*>(&Bs[bnb + i][bkk]) = p;   // merges to 2addr_b32
      }
    }
  };

  fetch(0);
  for (int k0 = 0; k0 < K; k0 += TILE_K) {
    stage();
    __syncthreads();

    // prefetch next K-step while WMMAs run below
    const int kn = k0 + TILE_K;
    if (kn < K) fetch(kn);

    // ---- A fragment: 16x32 bf16.  k(e) = (e>=8?16:0) + half*8 + (e&7) ----
    v16bf afrag = concat8(*reinterpret_cast<const v8bf*>(&As[mrow][half * 8]),
                          *reinterpret_cast<const v8bf*>(&As[mrow][16 + half * 8]));

    // ---- B fragments: 32x16 bf16. lane=col, k(e) = (e>=8?16:0) + half*8 + (e&7) ----
    #pragma unroll
    for (int t = 0; t < 4; ++t) {
      const int nr = t * 16 + l16;
      v16bf bfrag = concat8(*reinterpret_cast<const v8bf*>(&Bs[nr][half * 8]),
                            *reinterpret_cast<const v8bf*>(&Bs[nr][16 + half * 8]));
      acc[t] = __builtin_amdgcn_wmma_f32_16x16x32_bf16(false, afrag, false, bfrag,
                                                       (short)0, acc[t], false, false);
    }
    __syncthreads();
  }

  // ---- epilogue: C/D layout -> m = r + 8*half, n = lane&15 ----
  #pragma unroll
  for (int r = 0; r < 8; ++r) {
    const int gm = m0 + wave * 16 + half * 8 + r;
    if (gm >= M) continue;
    #pragma unroll
    for (int t = 0; t < 4; ++t) {
      const int gn = n0 + t * 16 + l16;
      if (gn >= N) continue;
      float v = acc[t][r] * alpha;
      if (bias) v += bias[gn];
      if (act == 1) v = 0.5f * v * (1.0f + erff(v * 0.70710678118654752f)); // exact GELU
      if (resz) v += resz[(long long)gm * N + gn];
      C[(long long)gm * N + gn] = v;
    }
  }
}

// ---------------------------------------------------------------------------
// LayerNorm: one 256-thread block per row of length D (strided rows supported)
// ---------------------------------------------------------------------------
__device__ __forceinline__ float warp_red_sum(float v) {
  #pragma unroll
  for (int o = 16; o > 0; o >>= 1) v += __shfl_xor(v, o, 32);
  return v;
}

__global__ __launch_bounds__(256)
void layernorm_kernel(const float* __restrict__ x, const float* __restrict__ g,
                      const float* __restrict__ b, float* __restrict__ out,
                      int D, long long in_stride, long long out_stride)
{
  __shared__ float red[8];
  const long long row = blockIdx.x;
  const float* xr = x + row * in_stride;
  float* orow = out + row * out_stride;
  const int tid = threadIdx.x, wave = tid >> 5, lane = tid & 31;

  float s = 0.f, ss = 0.f;
  for (int c = tid; c < D; c += 256) { float v = xr[c]; s += v; ss += v * v; }

  s = warp_red_sum(s);
  if (lane == 0) red[wave] = s;
  __syncthreads();
  float tot = 0.f;
  #pragma unroll
  for (int i = 0; i < 8; ++i) tot += red[i];
  __syncthreads();

  ss = warp_red_sum(ss);
  if (lane == 0) red[wave] = ss;
  __syncthreads();
  float tot2 = 0.f;
  #pragma unroll
  for (int i = 0; i < 8; ++i) tot2 += red[i];
  __syncthreads();

  const float mean = tot / (float)D;
  const float var  = tot2 / (float)D - mean * mean;
  const float inv  = rsqrtf(var + EPSc);
  for (int c = tid; c < D; c += 256)
    orow[c] = (xr[c] - mean) * inv * g[c] + b[c];
}

// ---------------------------------------------------------------------------
// Split qkv [B,S,H,3*DH] -> q,k,v each [B,H,S,DH]
// ---------------------------------------------------------------------------
__global__ __launch_bounds__(256)
void split_qkv_kernel(const float* __restrict__ qkv, float* __restrict__ q,
                      float* __restrict__ k, float* __restrict__ v)
{
  long long idx = (long long)blockIdx.x * 256 + threadIdx.x;
  const long long total = (long long)Bc * Hc * Sc * DHc;
  if (idx >= total) return;
  const int d = (int)(idx % DHc); long long t = idx / DHc;
  const int s = (int)(t % Sc);    t /= Sc;
  const int h = (int)(t % Hc);
  const int b = (int)(t / Hc);
  const long long src = ((long long)(b * Sc + s)) * (3 * Dc) + (long long)h * 3 * DHc + d;
  q[idx] = qkv[src];
  k[idx] = qkv[src + DHc];
  v[idx] = qkv[src + 2 * DHc];
}

// ---------------------------------------------------------------------------
// Softmax over row j with fused relative-position gather:
//   s[j] = content[j] + pos[bh, i, clamp(j-i,-W,W)+W]
// grid = (S, B*H), block = 256, S = 1024 -> 4 elems/thread. In-place.
// ---------------------------------------------------------------------------
__global__ __launch_bounds__(256)
void softmax_pos_kernel(float* __restrict__ scores, const float* __restrict__ pos)
{
  __shared__ float red[8];
  const int i  = blockIdx.x;
  const int bh = blockIdx.y;
  float* row = scores + ((long long)bh * Sc + i) * Sc;
  const float* prow = pos + ((long long)bh * Sc + i) * Rc;
  const int tid = threadIdx.x, wave = tid >> 5, lane = tid & 31;

  float vals[4];
  float m = -3.4e38f;
  #pragma unroll
  for (int t = 0; t < 4; ++t) {
    const int j = tid + t * 256;
    int off = j - i;
    off = off < -Wc ? -Wc : (off > Wc ? Wc : off);
    const float v = row[j] + prow[off + Wc];
    vals[t] = v;
    m = fmaxf(m, v);
  }
  #pragma unroll
  for (int o = 16; o > 0; o >>= 1) m = fmaxf(m, __shfl_xor(m, o, 32));
  if (lane == 0) red[wave] = m;
  __syncthreads();
  float bm = red[0];
  #pragma unroll
  for (int i2 = 1; i2 < 8; ++i2) bm = fmaxf(bm, red[i2]);
  __syncthreads();

  float sum = 0.f;
  #pragma unroll
  for (int t = 0; t < 4; ++t) { vals[t] = __expf(vals[t] - bm); sum += vals[t]; }
  sum = warp_red_sum(sum);
  if (lane == 0) red[wave] = sum;
  __syncthreads();
  float tot = 0.f;
  #pragma unroll
  for (int i2 = 0; i2 < 8; ++i2) tot += red[i2];
  __syncthreads();

  const float inv = 1.0f / tot;
  #pragma unroll
  for (int t = 0; t < 4; ++t) row[tid + t * 256] = vals[t] * inv;
}

// ---------------------------------------------------------------------------
// x[b,s,h*DH+d] += o[b,h,s,d]  (attention output transpose + residual)
// ---------------------------------------------------------------------------
__global__ __launch_bounds__(256)
void add_attn_out_kernel(float* __restrict__ x, const float* __restrict__ o)
{
  long long idx = (long long)blockIdx.x * 256 + threadIdx.x;
  const long long total = (long long)Bc * Hc * Sc * DHc;
  if (idx >= total) return;
  const int d = (int)(idx % DHc); long long t = idx / DHc;
  const int s = (int)(t % Sc);    t /= Sc;
  const int h = (int)(t % Hc);
  const int b = (int)(t / Hc);
  x[((long long)(b * Sc + s)) * Dc + h * DHc + d] += o[idx];
}

// ---------------------------------------------------------------------------
// Head: logits[b,o] = cls[b,:] @ head_w[:,o] + head_b[o]; mask==0 -> -inf
// ---------------------------------------------------------------------------
__global__ __launch_bounds__(256)
void head_mask_kernel(const float* __restrict__ cls, const float* __restrict__ w,
                      const float* __restrict__ bias, const int* __restrict__ mask,
                      float* __restrict__ out)
{
  const int idx = blockIdx.x * 256 + threadIdx.x;
  if (idx >= Bc * Oc) return;
  const int b = idx / Oc, o = idx % Oc;
  float acc = bias[o];
  for (int d = 0; d < Dc; ++d) acc += cls[b * Dc + d] * w[d * Oc + o];
  out[idx] = (mask[idx] == 0) ? -__builtin_huge_valf() : acc;
}

// ---------------------------------------------------------------------------
// Host orchestration
// ---------------------------------------------------------------------------
extern "C" void kernel_launch(void* const* d_in, const int* in_sizes, int n_in,
                              void* d_out, int out_size, void* d_ws, size_t ws_size,
                              hipStream_t stream)
{
  (void)in_sizes; (void)n_in; (void)out_size; (void)ws_size;

  const float* in_x      = (const float*)d_in[0];
  const float* ln1_g     = (const float*)d_in[1];
  const float* ln1_b     = (const float*)d_in[2];
  const float* qkv_w     = (const float*)d_in[3];
  const float* qkv_b     = (const float*)d_in[4];
  const float* rel_w     = (const float*)d_in[5];
  const float* rel_b     = (const float*)d_in[6];
  const float* ln2_g     = (const float*)d_in[7];
  const float* ln2_b     = (const float*)d_in[8];
  const float* mlp_w1    = (const float*)d_in[9];
  const float* mlp_b1    = (const float*)d_in[10];
  const float* mlp_w2    = (const float*)d_in[11];
  const float* mlp_b2    = (const float*)d_in[12];
  const float* normf_g   = (const float*)d_in[13];
  const float* normf_b   = (const float*)d_in[14];
  const float* head_w    = (const float*)d_in[15];
  const float* head_b    = (const float*)d_in[16];
  const int*   mask      = (const int*)d_in[17];

  // workspace carve-up (element counts)
  const long long nX   = (long long)Bc * Sc * Dc;        // 2,097,152
  const long long nQKV = (long long)Bc * Sc * 3 * Dc;    // 6,291,456
  const long long nHd  = (long long)Bc * Hc * Sc * DHc;  // 2,097,152
  const long long nPos = (long long)Bc * Hc * Sc * Rc;   // 4,292,608
  const long long nSc  = (long long)Bc * Hc * Sc * Sc;   // 33,554,432
  const long long nMh  = (long long)Bc * Sc * 4 * Dc;    // 8,388,608

  float* ws   = (float*)d_ws;
  float* X    = ws;            ws += nX;      // running residual stream
  float* Hbuf = ws;            ws += nX;      // LN output
  float* QKV  = ws;            ws += nQKV;
  float* Q    = ws;            ws += nHd;
  float* Kt   = ws;            ws += nHd;
  float* V    = ws;            ws += nHd;
  float* POS  = ws;            ws += nPos;
  float* SCO  = ws;            ws += nSc;     // scores / attn probs (in-place)
  float* AO   = ws;            ws += nHd;     // attn output [B,H,S,DH]
  float* MH   = ws;            ws += nMh;     // MLP hidden
  float* CLS  = ws;            ws += (long long)Bc * Dc;

  hipMemcpyAsync(X, in_x, (size_t)nX * sizeof(float), hipMemcpyDeviceToDevice, stream);

  const int   Mtok  = Bc * Sc;            // 4096 token rows
  const float scale = 0.125f;             // 1/sqrt(DH)
  const dim3 blk(256);

  auto gemm = [&](const float* A, const float* Bm, const float* bias, const float* res,
                  float* Cm, int M, int N, int K,
                  long long sA, long long sB, long long sC, int batch,
                  float alpha, int bT, int act) {
    dim3 grid((N + TILE_N - 1) / TILE_N, (M + TILE_M - 1) / TILE_M, batch);
    if (bT)
      wmma_gemm_kernel<1><<<grid, blk, 0, stream>>>(A, Bm, bias, res, Cm, M, N, K,
                                                    sA, sB, sC, alpha, act);
    else
      wmma_gemm_kernel<0><<<grid, blk, 0, stream>>>(A, Bm, bias, res, Cm, M, N, K,
                                                    sA, sB, sC, alpha, act);
  };

  for (int l = 0; l < Lc; ++l) {
    const float* l1g = ln1_g + (long long)l * Dc;
    const float* l1b = ln1_b + (long long)l * Dc;
    const float* qw  = qkv_w + (long long)l * Dc * 3 * Dc;
    const float* qb  = qkv_b + (long long)l * 3 * Dc;
    const float* l2g = ln2_g + (long long)l * Dc;
    const float* l2b = ln2_b + (long long)l * Dc;
    const float* w1  = mlp_w1 + (long long)l * Dc * 4 * Dc;
    const float* b1  = mlp_b1 + (long long)l * 4 * Dc;
    const float* w2  = mlp_w2 + (long long)l * 4 * Dc * Dc;
    const float* b2  = mlp_b2 + (long long)l * Dc;

    // h = LN1(x)
    layernorm_kernel<<<dim3(Mtok), blk, 0, stream>>>(X, l1g, l1b, Hbuf, Dc, Dc, Dc);
    // qkv = h @ qkv_w + qkv_b
    gemm(Hbuf, qw, qb, nullptr, QKV, Mtok, 3 * Dc, Dc, 0, 0, 0, 1, 1.0f, 0, 0);
    // split + transpose to [B,H,S,DH]
    split_qkv_kernel<<<dim3((unsigned)((nHd + 255) / 256)), blk, 0, stream>>>(QKV, Q, Kt, V);
    // pos = q @ rel_w + rel_b   ([B*H*S, 64] @ [64, 131])
    gemm(Q, rel_w, rel_b, nullptr, POS, Bc * Hc * Sc, Rc, DHc, 0, 0, 0, 1, 1.0f, 0, 0);
    // content = scale * Q @ K^T  (batched over B*H)
    gemm(Q, Kt, nullptr, nullptr, SCO, Sc, Sc, DHc,
         (long long)Sc * DHc, (long long)Sc * DHc, (long long)Sc * Sc,
         Bc * Hc, scale, 1, 0);
    // attn = softmax(content + gathered pos) (in-place)
    softmax_pos_kernel<<<dim3(Sc, Bc * Hc), blk, 0, stream>>>(SCO, POS);
    // out = attn @ V (batched)
    gemm(SCO, V, nullptr, nullptr, AO, Sc, DHc, Sc,
         (long long)Sc * Sc, (long long)Sc * DHc, (long long)Sc * DHc,
         Bc * Hc, 1.0f, 0, 0);
    // x += transpose(out)
    add_attn_out_kernel<<<dim3((unsigned)((nHd + 255) / 256)), blk, 0, stream>>>(X, AO);
    // h2 = LN2(x)
    layernorm_kernel<<<dim3(Mtok), blk, 0, stream>>>(X, l2g, l2b, Hbuf, Dc, Dc, Dc);
    // mh = gelu(h2 @ w1 + b1)
    gemm(Hbuf, w1, b1, nullptr, MH, Mtok, 4 * Dc, Dc, 0, 0, 0, 1, 1.0f, 0, 1);
    // x = x + mh @ w2 + b2
    gemm(MH, w2, b2, X, X, Mtok, Dc, 4 * Dc, 0, 0, 0, 1, 1.0f, 0, 0);
  }

  // final LN on CLS rows only (row stride S*D picks x[b,0,:])
  layernorm_kernel<<<dim3(Bc), blk, 0, stream>>>(X, normf_g, normf_b, CLS,
                                                 Dc, (long long)Sc * Dc, Dc);
  // logits + mask
  head_mask_kernel<<<dim3((Bc * Oc + 255) / 256), blk, 0, stream>>>(
      CLS, head_w, head_b, mask, (float*)d_out);
}
